// Net_72791105732854
// MI455X (gfx1250) — compile-verified, matrix operand order checked by reading, not run on hardware
//
#include <hip/hip_runtime.h>
#include <hip/hip_bf16.h>

// ---------------------------------------------------------------------------
// 2-layer GCN (Reddit-like) for gfx1250 / MI455X.
//   h0 = x @ W1                      (WMMA bf16 -> f32, 16x16x32)
//   h1 = relu(A_norm h0 + b1)        (edge-parallel f32 atomics + epilogue)
//   h2 = h1 @ W2                     (WMMA bf16 -> f32)
//   out = log_softmax(A_norm h2 + b2)
// Fragments are staged in LDS in fragment-contiguous order so each lane
// builds its v16bf operand with two ds_load_b128 ops.
// ---------------------------------------------------------------------------

#define F_IN 602
#define F_IN_PAD 608   // 19 * 32
#define NKT1 19        // K tiles of 32 for GEMM1
#define HID  64
#define NCLS 41
#define NCLS_PAD 48    // 3 * 16

typedef __attribute__((ext_vector_type(16))) __bf16          v16bf;
typedef __attribute__((ext_vector_type(8)))  float           v8f;
typedef __attribute__((ext_vector_type(8)))  unsigned short  u16x8;

struct bfpair { u16x8 lo; u16x8 hi; };

__device__ __forceinline__ unsigned short f2bf(float f) {
    unsigned u = __builtin_bit_cast(unsigned, f);
    u += 0x7FFFu + ((u >> 16) & 1u);          // round-to-nearest-even
    return (unsigned short)(u >> 16);
}

// Load one WMMA 16-bit operand fragment: two contiguous 16B runs, 32B apart.
// base must be 16B aligned.  elements e0..7 at base[0..7], e8..15 at base[16..23].
__device__ __forceinline__ v16bf load_frag(const unsigned short* base) {
    bfpair p;
    p.lo = *(const u16x8*)(base);
    p.hi = *(const u16x8*)(base + 16);
    return __builtin_bit_cast(v16bf, p);
}

// ---------------------------------------------------------------- degrees ---
__global__ void k_degree(const int* __restrict__ dst, float* __restrict__ deg, int E) {
    long long i = (long long)blockIdx.x * blockDim.x + threadIdx.x;
    if (i < E) atomicAdd(&deg[dst[i]], 1.0f);
}

__global__ void k_dinv(float* __restrict__ deg, int N) {
    long long i = (long long)blockIdx.x * blockDim.x + threadIdx.x;
    if (i < N) deg[i] = rsqrtf(deg[i] + 1.0f);   // +1 self-loop, deg >= 1
}

// -------------------------------------------------------------- weight pack -
// W1 -> bf16, K-tiled column-major: w1p[kt][n][kk], kt = k/32, kk = k%32.
__global__ void k_pack_w1(const float* __restrict__ W1, unsigned short* __restrict__ Wp) {
    int i = blockIdx.x * blockDim.x + threadIdx.x;          // [F_IN_PAD * HID]
    if (i >= F_IN_PAD * HID) return;
    int k = i / HID, n = i % HID;
    unsigned short v = (k < F_IN) ? f2bf(W1[k * HID + n]) : (unsigned short)0;
    Wp[(k >> 5) * (HID * 32) + n * 32 + (k & 31)] = v;
}

// W2 -> bf16, K-tiled column-major with N padded to 48: w2p[kt][n][kk]
__global__ void k_pack_w2(const float* __restrict__ W2, unsigned short* __restrict__ Wp) {
    int i = blockIdx.x * blockDim.x + threadIdx.x;          // [HID * NCLS_PAD]
    if (i >= HID * NCLS_PAD) return;
    int k = i / NCLS_PAD, n = i % NCLS_PAD;
    unsigned short v = (n < NCLS) ? f2bf(W2[k * NCLS + n]) : (unsigned short)0;
    Wp[(k >> 5) * (NCLS_PAD * 32) + n * 32 + (k & 31)] = v;
}

// ------------------------------------------------- GEMM1: h0 = x @ W1 -------
// 128 threads = 4 waves. Block computes 64 rows x 64 cols. Wave w owns rows
// [w*16, w*16+16) of the block tile; 4 WMMA N-tiles of 16.
__global__ void k_gemm1(const float* __restrict__ x,
                        const unsigned short* __restrict__ Wp,   // [NKT1][64][32] bf16
                        float* __restrict__ h0, int N) {
    __shared__ __attribute__((aligned(16))) unsigned short sA[64 * 32];  // row-major
    __shared__ __attribute__((aligned(16))) unsigned short sW[64 * 32];  // col-major

    const int tid  = threadIdx.x;
    const int wave = tid >> 5;
    const int lane = tid & 31;
    const int lo   = lane & 15;
    const int hi   = lane >> 4;
    const long long rowBase = (long long)blockIdx.x * 64;

    v8f acc[4];
    #pragma unroll
    for (int t = 0; t < 4; ++t)
        #pragma unroll
        for (int r = 0; r < 8; ++r) acc[t][r] = 0.0f;

    for (int kt = 0; kt < NKT1; ++kt) {
        const int k0 = kt * 32;
        __syncthreads();
        // stage A tile (f32 -> bf16), coalesced over k; kk is constant per thread
        for (int i = tid; i < 64 * 32; i += 128) {
            int r = i >> 5, kk = i & 31;
            long long row = rowBase + r;
            int k = k0 + kk;
            float v = (row < N && k < F_IN) ? x[row * F_IN + k] : 0.0f;
            sA[i] = f2bf(v);
            // prefetch next 128B K-chunk of this row (one thread per row)
            if (kk == 0 && row < N && k0 + 32 < F_IN)
                __builtin_prefetch(&x[row * F_IN + k0 + 32], 0, 3);
        }
        // stage W tile: already fragment-ordered in global, straight 4KB copy
        for (int i = tid; i < 64 * 32; i += 128)
            sW[i] = Wp[kt * (64 * 32) + i];
        __syncthreads();

        // A fragment: two ds_load_b128 per lane
        const v16bf a = load_frag(&sA[(wave * 16 + lo) * 32 + hi * 8]);
        #pragma unroll
        for (int nt = 0; nt < 4; ++nt) {
            const v16bf b = load_frag(&sW[(nt * 16 + lo) * 32 + hi * 8]);
            acc[nt] = __builtin_amdgcn_wmma_f32_16x16x32_bf16(
                false, a, false, b, (short)0, acc[nt], false, false);
        }
    }

    // C layout: VGPR r -> M = r + hi*8 ; N = lo (+16*nt)
    #pragma unroll
    for (int nt = 0; nt < 4; ++nt)
        #pragma unroll
        for (int r = 0; r < 8; ++r) {
            long long row = rowBase + wave * 16 + hi * 8 + r;
            if (row < N) h0[row * HID + nt * 16 + lo] = acc[nt][r];
        }
}

// ------------------------------------------------- GEMM2: h2 = h1 @ W2 ------
__global__ void k_gemm2(const float* __restrict__ h1,
                        const unsigned short* __restrict__ Wp,   // [2][48][32] bf16
                        float* __restrict__ h2, int N) {
    __shared__ __attribute__((aligned(16))) unsigned short sA[64 * 32];
    __shared__ __attribute__((aligned(16))) unsigned short sW[NCLS_PAD * 32];

    const int tid  = threadIdx.x;
    const int wave = tid >> 5;
    const int lane = tid & 31;
    const int lo   = lane & 15;
    const int hi   = lane >> 4;
    const long long rowBase = (long long)blockIdx.x * 64;

    v8f acc[3];
    #pragma unroll
    for (int t = 0; t < 3; ++t)
        #pragma unroll
        for (int r = 0; r < 8; ++r) acc[t][r] = 0.0f;

    #pragma unroll
    for (int kt = 0; kt < 2; ++kt) {
        const int k0 = kt * 32;
        __syncthreads();
        for (int i = tid; i < 64 * 32; i += 128) {
            int r = i >> 5, kk = i & 31;
            long long row = rowBase + r;
            float v = (row < N) ? h1[row * HID + k0 + kk] : 0.0f;
            sA[i] = f2bf(v);
        }
        for (int i = tid; i < NCLS_PAD * 32; i += 128)
            sW[i] = Wp[kt * (NCLS_PAD * 32) + i];
        __syncthreads();

        const v16bf a = load_frag(&sA[(wave * 16 + lo) * 32 + hi * 8]);
        #pragma unroll
        for (int nt = 0; nt < 3; ++nt) {
            const v16bf b = load_frag(&sW[(nt * 16 + lo) * 32 + hi * 8]);
            acc[nt] = __builtin_amdgcn_wmma_f32_16x16x32_bf16(
                false, a, false, b, (short)0, acc[nt], false, false);
        }
    }

    #pragma unroll
    for (int nt = 0; nt < 3; ++nt)
        #pragma unroll
        for (int r = 0; r < 8; ++r) {
            long long row = rowBase + wave * 16 + hi * 8 + r;
            if (row < N) h2[row * NCLS_PAD + nt * 16 + lo] = acc[nt][r];
        }
}

// ---------------------------------------------- SpMM 1 (edge scatter, F=64) -
// thread t handles edge t/8, features [8*(t%8), +8)
__global__ void k_edge1(const int* __restrict__ src, const int* __restrict__ dst,
                        const float* __restrict__ dinv, const float* __restrict__ h0,
                        float* __restrict__ agg, int E) {
    long long t = (long long)blockIdx.x * blockDim.x + threadIdx.x;
    if (t >= (long long)E * 8) return;
    int e  = (int)(t >> 3);
    int f0 = (int)(t & 7) * 8;
    int s = src[e], d = dst[e];
    float w = dinv[s] * dinv[d];
    const float4* hp = (const float4*)(h0 + (long long)s * HID + f0);
    float4 v0 = hp[0], v1 = hp[1];
    float* op = agg + (long long)d * HID + f0;
    atomicAdd(op + 0, w * v0.x); atomicAdd(op + 1, w * v0.y);
    atomicAdd(op + 2, w * v0.z); atomicAdd(op + 3, w * v0.w);
    atomicAdd(op + 4, w * v1.x); atomicAdd(op + 5, w * v1.y);
    atomicAdd(op + 6, w * v1.z); atomicAdd(op + 7, w * v1.w);
}

// self-loop + bias + relu, in-place on agg -> h1
__global__ void k_epilogue1(float* __restrict__ agg, const float* __restrict__ h0,
                            const float* __restrict__ dinv, const float* __restrict__ b1,
                            int N) {
    long long i = (long long)blockIdx.x * blockDim.x + threadIdx.x;
    if (i >= (long long)N * HID) return;
    int n = (int)(i >> 6);
    int f = (int)(i & 63);
    float di = dinv[n];
    float v = agg[i] + di * di * h0[i] + b1[f];
    agg[i] = v > 0.0f ? v : 0.0f;
}

// ---------------------------------------------- SpMM 2 (edge scatter, F=41) -
__global__ void k_edge2(const int* __restrict__ src, const int* __restrict__ dst,
                        const float* __restrict__ dinv, const float* __restrict__ h2,
                        float* __restrict__ agg, int E) {
    long long t = (long long)blockIdx.x * blockDim.x + threadIdx.x;
    if (t >= (long long)E * NCLS) return;
    int e = (int)(t / NCLS);
    int c = (int)(t % NCLS);
    int s = src[e], d = dst[e];
    float w = dinv[s] * dinv[d];
    atomicAdd(&agg[(long long)d * NCLS + c], w * h2[(long long)s * NCLS_PAD + c]);
}

// ------------------------------- self-loop + bias + log_softmax (per node) --
__global__ void k_final(const float* __restrict__ agg, const float* __restrict__ h2,
                        const float* __restrict__ dinv, const float* __restrict__ b2,
                        float* __restrict__ out, int N) {
    long long n = (long long)blockIdx.x * blockDim.x + threadIdx.x;
    if (n >= N) return;
    float di = dinv[n];
    float wii = di * di;
    float v[NCLS];
    float mx = -3.402823e38f;
    #pragma unroll
    for (int c = 0; c < NCLS; ++c) {
        float t = agg[n * NCLS + c] + wii * h2[n * NCLS_PAD + c] + b2[c];
        v[c] = t;
        mx = fmaxf(mx, t);
    }
    float s = 0.0f;
    #pragma unroll
    for (int c = 0; c < NCLS; ++c) s += __expf(v[c] - mx);
    float lse = __logf(s) + mx;
    #pragma unroll
    for (int c = 0; c < NCLS; ++c) out[n * NCLS + c] = v[c] - lse;
}

// ---------------------------------------------------------------------------
extern "C" void kernel_launch(void* const* d_in, const int* in_sizes, int n_in,
                              void* d_out, int out_size, void* d_ws, size_t ws_size,
                              hipStream_t stream) {
    const float* x   = (const float*)d_in[0];
    const int*   src = (const int*)  d_in[1];
    const int*   dst = (const int*)  d_in[2];
    const float* W1  = (const float*)d_in[3];
    const float* b1  = (const float*)d_in[4];
    const float* W2  = (const float*)d_in[5];
    const float* b2  = (const float*)d_in[6];
    float* out = (float*)d_out;

    const int N = in_sizes[0] / F_IN;
    const int E = in_sizes[1];

    // ---- workspace layout (256B aligned) ----
    size_t off = 0;
    auto take = [&](size_t bytes) {
        off = (off + 255) & ~(size_t)255;
        size_t o = off; off += bytes; return o;
    };
    char* ws = (char*)d_ws;
    float* dinv = (float*)(ws + take((size_t)N * 4));                  // deg -> dinv in place
    float* h0   = (float*)(ws + take((size_t)N * HID * 4));            // later reused as agg2
    float* agg1 = (float*)(ws + take((size_t)N * HID * 4));            // becomes h1 in place
    float* h2   = (float*)(ws + take((size_t)N * NCLS_PAD * 4));
    unsigned short* w1p = (unsigned short*)(ws + take((size_t)F_IN_PAD * HID * 2));
    unsigned short* w2p = (unsigned short*)(ws + take((size_t)HID * NCLS_PAD * 2));
    float* agg2 = h0;  // h0 is dead after epilogue1; N*41*4 <= N*64*4

    // ---- degrees / normalization ----
    hipMemsetAsync(dinv, 0, (size_t)N * 4, stream);
    k_degree<<<(E + 255) / 256, 256, 0, stream>>>(dst, dinv, E);
    k_dinv<<<(N + 255) / 256, 256, 0, stream>>>(dinv, N);

    // ---- pack weights to bf16, fragment-tiled ----
    k_pack_w1<<<(F_IN_PAD * HID + 255) / 256, 256, 0, stream>>>(W1, w1p);
    k_pack_w2<<<(HID * NCLS_PAD + 255) / 256, 256, 0, stream>>>(W2, w2p);

    // ---- layer 1 ----
    hipMemsetAsync(agg1, 0, (size_t)N * HID * 4, stream);
    k_gemm1<<<(N + 63) / 64, 128, 0, stream>>>(x, w1p, h0, N);
    {
        long long th = (long long)E * 8;
        k_edge1<<<(unsigned)((th + 255) / 256), 256, 0, stream>>>(src, dst, dinv, h0, agg1, E);
    }
    {
        long long th = (long long)N * HID;
        k_epilogue1<<<(unsigned)((th + 255) / 256), 256, 0, stream>>>(agg1, h0, dinv, b1, N);
    }

    // ---- layer 2 ----
    k_gemm2<<<(N + 63) / 64, 128, 0, stream>>>(agg1 /* = h1 */, w2p, h2, N);
    hipMemsetAsync(agg2, 0, (size_t)N * NCLS * 4, stream);
    {
        long long th = (long long)E * NCLS;
        k_edge2<<<(unsigned)((th + 255) / 256), 256, 0, stream>>>(src, dst, dinv, h2, agg2, E);
    }
    k_final<<<(N + 255) / 256, 256, 0, stream>>>(agg2, h2, dinv, b2, out, N);

    (void)n_in; (void)out_size; (void)ws_size;
}
